// PureCartesianSparseTransformerLayer_66881230733947
// MI455X (gfx1250) — compile-verified, matrix operand order checked by reading, not run on hardware
//
#include <hip/hip_runtime.h>
#include <hip/hip_bf16.h>

typedef __attribute__((ext_vector_type(16))) _Float16 v16h;
typedef __attribute__((ext_vector_type(8)))  _Float16 h8;
typedef __attribute__((ext_vector_type(8)))  float    v8f;

static constexpr int N_NODES = 10000;
static constexpr int N_EDGES = 100000;
static constexpr int EPAD    = 101600;   // E + 100*16 slack for tile-aligned buckets
static constexpr int OUTF    = 208;      // 16 + 48 + 144
static constexpr int MAIN_BLOCKS = 1588; // 4 waves/block -> 6352 tiles >= EPAD/16

// workspace layout (bytes)
static constexpr size_t OFF_AI   = 0;        // 80 floats  : Ai table 10x8
static constexpr size_t OFF_B3F  = 512;      // 4800 floats: folded bias 100x48
static constexpr size_t OFF_W1   = 19712;    // 2048 halves: fc_w1 B-pack
static constexpr size_t OFF_W2   = 23808;    // 4096 halves: fc_w2 B-pack
static constexpr size_t OFF_W3   = 32000;    // 307200 halves: folded fc_w3 B-pack
static constexpr size_t OFF_PAIR = 646400;   // 100000 int
static constexpr size_t OFF_CNTD = 1046400;  // 10000 int
static constexpr size_t OFF_BINC = 1086400;  // 100 int
static constexpr size_t OFF_BINS = 1086816;  // 101 int
static constexpr size_t OFF_CUR  = 1087232;  // 100 int
static constexpr size_t OFF_SORT = 1087744;  // 101600 int

// fast silu: x * v_rcp_f32(1 + e^-x)  (no IEEE div chain)
__device__ inline float siluf(float x) {
  return x * __builtin_amdgcn_rcpf(1.0f + __expf(-x));
}
// precise-ish silu for the tiny one-shot Ai table (cost irrelevant)
__device__ inline float siluf_precise(float x) { return x / (1.0f + __expf(-x)); }

// K-index inside a 32-wide K block for B-operand layout (mirror of ISA 16-bit A layout)
__device__ inline int bkidx(int lane, int j) {
  int h = (lane & 16) ? 8 : 0;
  return (j < 8) ? (h + j) : (16 + h + (j - 8));
}

__device__ inline v8f wmma16(v16h a, v16h b, v8f c) {
  return __builtin_amdgcn_wmma_f32_16x16x32_f16(false, a, false, b, (short)0, c, false, false);
}

// ---------------------------------------------------------------- init / zero
__global__ void k_zero(float* __restrict__ out, int* __restrict__ cntDst,
                       int* __restrict__ binCnt, int* __restrict__ sortedIdx) {
  int i = blockIdx.x * blockDim.x + threadIdx.x;
  if (i < N_NODES * OUTF) out[i] = 0.0f;
  if (i < N_NODES) cntDst[i] = 0;
  if (i < 100) binCnt[i] = 0;
  if (i < EPAD) sortedIdx[i] = -1;
}

// ------------------------------------------- Ai table + pack fc_w1/fc_w2 (f16)
__global__ void k_ai(const float* __restrict__ embT, const float* __restrict__ mw1,
                     const float* __restrict__ mb1, const float* __restrict__ mw2,
                     const float* __restrict__ mb2, const float* __restrict__ fw1,
                     const float* __restrict__ fw2, float* __restrict__ AiTab,
                     _Float16* __restrict__ w1p, _Float16* __restrict__ w2p) {
  int t = threadIdx.x;
  if (t < 80) {  // Ai[z, s] = silu(emb[z] @ w1 + b1) @ w2 + b2
    int z = t >> 3, s = t & 7;
    float acc = mb2[s];
    for (int k = 0; k < 64; ++k) {
      float h = mb1[k];
      for (int i = 0; i < 16; ++i) h += embT[z * 16 + i] * mw1[i * 64 + k];
      acc += siluf_precise(h) * mw2[k * 8 + s];
    }
    AiTab[z * 8 + s] = acc;
  }
  // fc_w1: [16(pad32) x 64] -> 4 N-blocks, layout [nb][lane][16]
  for (int idx = t; idx < 2048; idx += blockDim.x) {
    int nb = idx >> 9, lane = (idx >> 4) & 31, j = idx & 15;
    int n = lane & 15;
    int k = bkidx(lane, j);
    float v = (k < 16) ? fw1[k * 64 + nb * 16 + n] : 0.0f;
    w1p[idx] = (_Float16)v;
  }
  // fc_w2: [64 x 64] -> 2 K-blocks x 4 N-blocks, layout [kb][nb][lane][16]
  for (int idx = t; idx < 4096; idx += blockDim.x) {
    int kb = idx >> 11, nb = (idx >> 9) & 3, lane = (idx >> 4) & 31, j = idx & 15;
    int n = lane & 15;
    int k = kb * 32 + bkidx(lane, j);
    w2p[idx] = (_Float16)fw2[k * 64 + nb * 16 + n];
  }
}

// ---------------------------------------------------------- per-edge pair ids
__global__ void k_pairs(const int* __restrict__ A, const int* __restrict__ esrc,
                        const int* __restrict__ edst, int* __restrict__ pairId,
                        int* __restrict__ binCnt, int* __restrict__ cntDst) {
  int e = blockIdx.x * blockDim.x + threadIdx.x;
  if (e >= N_EDGES) return;
  int p = A[esrc[e]] * 10 + A[edst[e]];
  pairId[e] = p;
  atomicAdd(&binCnt[p], 1);
  atomicAdd(&cntDst[edst[e]], 1);
}

// ---------------------------------------------- padded scan (tile-aligned bins)
__global__ void k_scan(const int* __restrict__ binCnt, int* __restrict__ binStart,
                       int* __restrict__ cursor) {
  if (threadIdx.x == 0 && blockIdx.x == 0) {
    int acc = 0;
    for (int p = 0; p < 100; ++p) {
      binStart[p] = acc;
      cursor[p] = acc;
      acc += ((binCnt[p] + 15) >> 4) << 4;
    }
    binStart[100] = acc;
  }
}

__global__ void k_scatter(const int* __restrict__ pairId, int* __restrict__ cursor,
                          int* __restrict__ sortedIdx) {
  int e = blockIdx.x * blockDim.x + threadIdx.x;
  if (e >= N_EDGES) return;
  int pos = atomicAdd(&cursor[pairId[e]], 1);
  sortedIdx[pos] = e;
}

// --------------------------- fold fc_w3 / fc_b3 with Ai outer product per pair
__global__ __launch_bounds__(256) void k_fold(const float* __restrict__ fw3,
                                              const float* __restrict__ fb3,
                                              const float* __restrict__ AiTab,
                                              _Float16* __restrict__ w3p,
                                              float* __restrict__ b3f) {
  int p = blockIdx.x;
  int kbnb = blockIdx.y;
  int kb = kbnb / 3, nb = kbnb % 3;
  int zs = p / 10, zd = p % 10;
  __shared__ float As[8], Ad[8];
  if (threadIdx.x < 8) As[threadIdx.x] = AiTab[zs * 8 + threadIdx.x];
  else if (threadIdx.x < 16) Ad[threadIdx.x - 8] = AiTab[zd * 8 + (threadIdx.x - 8)];
  __syncthreads();

  for (int t = threadIdx.x; t < 512; t += 256) {
    int lane = t >> 4, j = t & 15;
    int k = kb * 32 + bkidx(lane, j);
    int lo = nb * 16 + (lane & 15);
    const float* wr = fw3 + (size_t)k * 3072 + lo * 64;
    float sum = 0.0f;
#pragma unroll
    for (int a = 0; a < 8; ++a) {
      float fa = As[a];
#pragma unroll
      for (int b = 0; b < 8; ++b) sum += wr[a * 8 + b] * Ad[b] * fa;
    }
    w3p[(((size_t)(p * 2 + kb) * 3 + nb) * 32 + lane) * 16 + j] = (_Float16)sum;
  }
  if (kbnb == 0 && threadIdx.x < 48) {
    int lo = threadIdx.x;
    const float* br = fb3 + lo * 64;
    float sum = 0.0f;
    for (int a = 0; a < 8; ++a)
      for (int b = 0; b < 8; ++b) sum += br[a * 8 + b] * As[a] * Ad[b];
    b3f[p * 48 + lo] = sum;
  }
}

// --------------------------------------------------------------- main kernel
__device__ inline v16h loadA(const _Float16* row, int kb, int half) {
  const h8 lo = *(const h8*)(row + kb * 32 + half * 8);
  const h8 hi = *(const h8*)(row + kb * 32 + 16 + half * 8);
  v16h a;
#pragma unroll
  for (int i = 0; i < 8; ++i) { a[i] = lo[i]; a[i + 8] = hi[i]; }
  return a;
}

__global__ __launch_bounds__(128) void k_main(
    const float* __restrict__ pos, const int* __restrict__ batch,
    const int* __restrict__ esrc, const int* __restrict__ edst,
    const float* __restrict__ shifts, const float* __restrict__ cell,
    const int* __restrict__ pairId, const int* __restrict__ sortedIdx,
    const _Float16* __restrict__ w1p, const float* __restrict__ fb1,
    const _Float16* __restrict__ w2p, const float* __restrict__ fb2,
    const _Float16* __restrict__ w3p, const float* __restrict__ b3f,
    float* __restrict__ out) {
  __shared__ _Float16 hbuf[4][16 * 64];
  __shared__ int      eD[4][16];
  __shared__ float    nv[4][16][3];

  const int w    = threadIdx.x >> 5;
  const int lane = threadIdx.x & 31;
  const int l16  = lane & 15;
  const int half = lane >> 4;
  const int tile = blockIdx.x * 4 + w;

  int sIdx = tile * 16 + l16;
  int e = (sIdx < EPAD) ? sortedIdx[sIdx] : -1;
  bool valid = (e >= 0);

  float r = 0.0f, nx = 0.0f, ny = 0.0f, nz = 0.0f;
  int dn = -1, pp = 0;
  if (valid) {
    int s = esrc[e], d = edst[e];
    dn = d;
    pp = pairId[e];
    int g = batch[s];
    float s0 = shifts[e * 3 + 0], s1 = shifts[e * 3 + 1], s2 = shifts[e * 3 + 2];
    const float* cg = cell + g * 9;
    float sh0 = s0 * cg[0] + s1 * cg[3] + s2 * cg[6];
    float sh1 = s0 * cg[1] + s1 * cg[4] + s2 * cg[7];
    float sh2 = s0 * cg[2] + s1 * cg[5] + s2 * cg[8];
    float vx = pos[d * 3 + 0] - pos[s * 3 + 0] + sh0;
    float vy = pos[d * 3 + 1] - pos[s * 3 + 1] + sh1;
    float vz = pos[d * 3 + 2] - pos[s * 3 + 2] + sh2;
    float s2n = vx * vx + vy * vy + vz * vz;
    r = __builtin_amdgcn_sqrtf(s2n);
    float ir = __builtin_amdgcn_rcpf(fmaxf(r, 1e-9f));
    nx = vx * ir; ny = vy * ir; nz = vz * ir;
  }
  if (half == 0) { eD[w][l16] = dn; nv[w][l16][0] = nx; nv[w][l16][1] = ny; nv[w][l16][2] = nz; }

  // broadcast bucket pair id from lane 0 (buckets are tile-aligned)
  int pair = __shfl(pp, 0, 32);
  pair = (pair < 0) ? 0 : ((pair > 99) ? 99 : pair);

  // radial basis directly in A-operand layout (K=16 real + 16 zero pad)
  v16h a1;
  const float step = 4.0f / 17.0f, istep = 17.0f / 4.0f;
#pragma unroll
  for (int i = 0; i < 8; ++i) {
    int k = half * 8 + i;
    float dv = (r - (float)(k + 1) * step) * istep;
    float ev = valid ? __expf(-dv * dv) * 3.5714285714f : 0.0f;
    a1[i] = (_Float16)ev;
    a1[i + 8] = (_Float16)0.0f;
  }

  // layer 1: [16x32] x [32x64]
  const v16h* W1 = (const v16h*)w1p;
  v8f acc1[4];
#pragma unroll
  for (int nb = 0; nb < 4; ++nb) {
    v8f c = {};
    acc1[nb] = wmma16(a1, W1[nb * 32 + lane], c);
  }
#pragma unroll
  for (int nb = 0; nb < 4; ++nb) {
    float bv = fb1[nb * 16 + l16];
#pragma unroll
    for (int rr = 0; rr < 8; ++rr)
      hbuf[w][(rr + 8 * half) * 64 + nb * 16 + l16] = (_Float16)siluf(acc1[nb][rr] + bv);
  }
  __syncthreads();

  const _Float16* myrow = &hbuf[w][l16 * 64];
  v16h a2[2];
#pragma unroll
  for (int kb = 0; kb < 2; ++kb) a2[kb] = loadA(myrow, kb, half);
  __syncthreads();

  // layer 2: [16x64] x [64x64]
  const v16h* W2 = (const v16h*)w2p;
  v8f acc2[4];
#pragma unroll
  for (int nb = 0; nb < 4; ++nb) {
    v8f c = {};
    c = wmma16(a2[0], W2[(0 * 4 + nb) * 32 + lane], c);
    c = wmma16(a2[1], W2[(1 * 4 + nb) * 32 + lane], c);
    acc2[nb] = c;
  }
#pragma unroll
  for (int nb = 0; nb < 4; ++nb) {
    float bv = fb2[nb * 16 + l16];
#pragma unroll
    for (int rr = 0; rr < 8; ++rr)
      hbuf[w][(rr + 8 * half) * 64 + nb * 16 + l16] = (_Float16)siluf(acc2[nb][rr] + bv);
  }
  __syncthreads();

  v16h a3[2];
#pragma unroll
  for (int kb = 0; kb < 2; ++kb) a3[kb] = loadA(myrow, kb, half);

  // layer 3 (pair-folded): [16x64] x [64x48] -> m[e, l, o]; N-block nb == l
  const v16h* W3 = (const v16h*)w3p;
  v8f  c3[3];
  float bias[3];
#pragma unroll
  for (int nb = 0; nb < 3; ++nb) {
    v8f c = {};
    c = wmma16(a3[0], W3[((pair * 2 + 0) * 3 + nb) * 32 + lane], c);
    c = wmma16(a3[1], W3[((pair * 2 + 1) * 3 + nb) * 32 + lane], c);
    c3[nb] = c;
    bias[nb] = b3f[pair * 48 + nb * 16 + l16];
  }

  // scatter: one pass per row, dst/n loaded once
#pragma unroll
  for (int rr = 0; rr < 8; ++rr) {
    int row = rr + 8 * half;
    int d = eD[w][row];
    if (d < 0) continue;
    float wx = nv[w][row][0], wy = nv[w][row][1], wz = nv[w][row][2];
    float m0 = c3[0][rr] + bias[0];
    float m1 = c3[1][rr] + bias[1];
    float m2 = c3[2][rr] + bias[2];
    float* ob = out + (size_t)d * OUTF;
    atomicAdd(ob + l16, m0);
    float* p1 = ob + 16 + l16 * 3;
    atomicAdd(p1 + 0, m1 * wx);
    atomicAdd(p1 + 1, m1 * wy);
    atomicAdd(p1 + 2, m1 * wz);
    float tx = m2 * wx, ty = m2 * wy, tz = m2 * wz;
    float* p2 = ob + 64 + l16 * 9;
    atomicAdd(p2 + 0, tx * wx);
    atomicAdd(p2 + 1, tx * wy);
    atomicAdd(p2 + 2, tx * wz);
    atomicAdd(p2 + 3, ty * wx);
    atomicAdd(p2 + 4, ty * wy);
    atomicAdd(p2 + 5, ty * wz);
    atomicAdd(p2 + 6, tz * wx);
    atomicAdd(p2 + 7, tz * wy);
    atomicAdd(p2 + 8, tz * wz);
  }
}

// ---------------------------------------------------------------- mean divide
__global__ void k_div(float* __restrict__ out, const int* __restrict__ cntDst) {
  int i = blockIdx.x * blockDim.x + threadIdx.x;
  if (i >= N_NODES * OUTF) return;
  float c = (float)cntDst[i / OUTF];
  out[i] = out[i] / fmaxf(c, 1.0f);
}

// ----------------------------------------------------------------------------
extern "C" void kernel_launch(void* const* d_in, const int* in_sizes, int n_in,
                              void* d_out, int out_size, void* d_ws, size_t ws_size,
                              hipStream_t stream) {
  const float* pos    = (const float*)d_in[0];
  const int*   A      = (const int*)d_in[1];
  const int*   batch  = (const int*)d_in[2];
  const int*   esrc   = (const int*)d_in[3];
  const int*   edst   = (const int*)d_in[4];
  const float* shifts = (const float*)d_in[5];
  const float* cell   = (const float*)d_in[6];
  const float* embT   = (const float*)d_in[7];
  const float* mw1    = (const float*)d_in[8];
  const float* mb1    = (const float*)d_in[9];
  const float* mw2    = (const float*)d_in[10];
  const float* mb2    = (const float*)d_in[11];
  const float* fw1    = (const float*)d_in[12];
  const float* fb1    = (const float*)d_in[13];
  const float* fw2    = (const float*)d_in[14];
  const float* fb2    = (const float*)d_in[15];
  const float* fw3    = (const float*)d_in[16];
  const float* fb3    = (const float*)d_in[17];

  char* ws = (char*)d_ws;
  float*    AiTab   = (float*)(ws + OFF_AI);
  float*    b3f     = (float*)(ws + OFF_B3F);
  _Float16* w1p     = (_Float16*)(ws + OFF_W1);
  _Float16* w2p     = (_Float16*)(ws + OFF_W2);
  _Float16* w3p     = (_Float16*)(ws + OFF_W3);
  int*      pairId  = (int*)(ws + OFF_PAIR);
  int*      cntDst  = (int*)(ws + OFF_CNTD);
  int*      binCnt  = (int*)(ws + OFF_BINC);
  int*      binSt   = (int*)(ws + OFF_BINS);
  int*      cursor  = (int*)(ws + OFF_CUR);
  int*      sorted  = (int*)(ws + OFF_SORT);
  float*    out     = (float*)d_out;

  int zgrid = (N_NODES * OUTF + 255) / 256;
  int egrid = (N_EDGES + 255) / 256;

  k_zero<<<zgrid, 256, 0, stream>>>(out, cntDst, binCnt, sorted);
  k_ai<<<1, 256, 0, stream>>>(embT, mw1, mb1, mw2, mb2, fw1, fw2, AiTab, w1p, w2p);
  k_pairs<<<egrid, 256, 0, stream>>>(A, esrc, edst, pairId, binCnt, cntDst);
  k_scan<<<1, 32, 0, stream>>>(binCnt, binSt, cursor);
  k_scatter<<<egrid, 256, 0, stream>>>(pairId, cursor, sorted);
  k_fold<<<dim3(100, 6), 256, 0, stream>>>(fw3, fb3, AiTab, w3p, b3f);
  k_main<<<MAIN_BLOCKS, 128, 0, stream>>>(pos, batch, esrc, edst, shifts, cell,
                                          pairId, sorted, w1p, fb1, w2p, fb2,
                                          w3p, b3f, out);
  k_div<<<zgrid, 256, 0, stream>>>(out, cntDst);
}
